// SelfAttention_65214783422583
// MI455X (gfx1250) — matrix operand order repaired and checked by reading, not compiled, not run on hardware
//
#include <hip/hip_runtime.h>
#include <hip/hip_bf16.h>

typedef __bf16 bf16;
typedef __attribute__((ext_vector_type(16))) __bf16 v16bf;
typedef __attribute__((ext_vector_type(4)))  __bf16 v4bf;
typedef __attribute__((ext_vector_type(8)))  float  v8f;
typedef __attribute__((ext_vector_type(4)))  float  v4f;
typedef __attribute__((ext_vector_type(4)))  unsigned int u32x4;  // 16 bytes
typedef __attribute__((ext_vector_type(4)))  int v4i;

#define DIM    1024
#define HEADS  16
#define DHEAD  64
#define BATCH  4
#define SEQ    2048
#define ROWS   (BATCH * SEQ)   // 8192

// ---------------------------------------------------------------------------
// CDNA5 async global->LDS staging (guarded: falls back to sync path if the
// toolchain doesn't declare the gfx1250 builtins). The builtin's param 0 is
// 'int4 addrspace(1)*' (clang prints AS1 as '__device__'), so the casts below
// go through uintptr_t to the exact vector-pointer types.
// ---------------------------------------------------------------------------
#if defined(__has_builtin)
#if __has_builtin(__builtin_amdgcn_global_load_async_to_lds_b128) && \
    __has_builtin(__builtin_amdgcn_s_wait_asynccnt)
#define USE_ASYNC_LDS 1
#endif
#endif
#ifndef USE_ASYNC_LDS
#define USE_ASYNC_LDS 0
#endif

#if USE_ASYNC_LDS
typedef __attribute__((address_space(1))) v4i gv4i;   // global int4
typedef __attribute__((address_space(3))) v4i lv4i;   // LDS int4
static __device__ __forceinline__ gv4i* as_g(const void* p) {
    return (gv4i*)(unsigned long long)(uintptr_t)p;   // 64-bit global addr
}
static __device__ __forceinline__ lv4i* as_l(void* p) {
    return (lv4i*)(unsigned int)(uintptr_t)p;         // low 32 bits = LDS offset
}
#define ASYNC_CP16(gp, lp) \
    __builtin_amdgcn_global_load_async_to_lds_b128(as_g(gp), as_l(lp), 0, 0)
#define ASYNC_WAIT() __builtin_amdgcn_s_wait_asynccnt(0)
#else
#define ASYNC_CP16(gp, lp) (*(u32x4*)(lp) = *(const u32x4*)(gp))
#define ASYNC_WAIT() ((void)0)
#endif

// ---------------------------------------------------------------------------
// float -> bf16 conversion (4-wide)
// ---------------------------------------------------------------------------
__global__ void __launch_bounds__(256) cvt_f32_bf16(const v4f* __restrict__ in,
                                                    v4bf* __restrict__ out, int n4) {
    int i = blockIdx.x * blockDim.x + threadIdx.x;
    if (i < n4) {
        v4f a = in[i];
        v4bf o;
        o[0] = (bf16)a[0]; o[1] = (bf16)a[1];
        o[2] = (bf16)a[2]; o[3] = (bf16)a[3];
        out[i] = o;
    }
}

// ---------------------------------------------------------------------------
// C[M,N] = A[M,K](bf16,row-major) * B[K,N](bf16,row-major), OutT = bf16 or f32
// Block tile 128x128, K-tile 64, 256 threads = 8 waves (2x4), wave tile 64x32.
// A tile staged via async LDS DMA; B tile transposed on store.
// ---------------------------------------------------------------------------
__device__ inline void storeC(bf16* p, float v)  { *p = (bf16)v; }
__device__ inline void storeC(float* p, float v) { *p = v; }

template <typename OutT>
__global__ void __launch_bounds__(256) gemm_bf16(const bf16* __restrict__ A,
                                                 const bf16* __restrict__ B,
                                                 OutT* __restrict__ C,
                                                 int M, int N, int K) {
    __shared__ __align__(16) bf16 Asl[128 * 64];   // [row][k]
    __shared__ __align__(16) bf16 Btl[128 * 64];   // [n][k] (transposed)

    const int tid  = threadIdx.x;
    const int lane = tid & 31;
    const int wid  = tid >> 5;       // 0..7
    const int wm   = wid >> 2;       // 0..1 -> 64-row strip
    const int wn   = wid & 3;        // 0..3 -> 32-col strip
    const int half = lane >> 4;      // 0/1
    const int l16  = lane & 15;

    const int nBlocksN = N / 128;
    const int m0 = (blockIdx.x / nBlocksN) * 128;
    const int n0 = (blockIdx.x % nBlocksN) * 128;

    v8f acc[4][2];
    #pragma unroll
    for (int mt = 0; mt < 4; ++mt)
        #pragma unroll
        for (int nt = 0; nt < 2; ++nt)
            acc[mt][nt] = (v8f){0.f,0.f,0.f,0.f,0.f,0.f,0.f,0.f};

    for (int k0 = 0; k0 < K; k0 += 64) {
        __syncthreads();
        // A tile 128x64: 1024 chunks of 8 bf16, async DMA straight into LDS
        #pragma unroll
        for (int i = 0; i < 4; ++i) {
            int c = tid + i * 256;
            int row = c >> 3, cc = (c & 7) << 3;
            ASYNC_CP16(A + (size_t)(m0 + row) * K + k0 + cc, Asl + row * 64 + cc);
        }
        // B tile 64x128, stored transposed as [n][k]
        #pragma unroll
        for (int i = 0; i < 4; ++i) {
            int c = tid + i * 256;
            int kr = c >> 4, nc = (c & 15) << 3;
            u32x4 d = *(const u32x4*)(B + (size_t)(k0 + kr) * N + n0 + nc);
            const bf16* de = (const bf16*)&d;
            #pragma unroll
            for (int e = 0; e < 8; ++e) Btl[(nc + e) * 64 + kr] = de[e];
        }
        ASYNC_WAIT();
        __syncthreads();

        #pragma unroll
        for (int s = 0; s < 2; ++s) {
            // A fragments: lane row l16; e0..7 at k=8*half, e8..15 at k=16+8*half
            v16bf af[4];
            #pragma unroll
            for (int mt = 0; mt < 4; ++mt) {
                const bf16* p = Asl + (wm * 64 + mt * 16 + l16) * 64 + s * 32 + 8 * half;
                union { v16bf v; u32x4 q[2]; } u;
                u.q[0] = *(const u32x4*)p;
                u.q[1] = *(const u32x4*)(p + 16);
                af[mt] = u.v;
            }
            // B fragments: lane col n; e0..7 at k=8*half, e8..15 at k=16+8*half
            v16bf bfx[2];
            #pragma unroll
            for (int nt = 0; nt < 2; ++nt) {
                const bf16* p = Btl + (wn * 32 + nt * 16 + l16) * 64 + s * 32 + 8 * half;
                union { v16bf v; u32x4 q[2]; } u;
                u.q[0] = *(const u32x4*)p;
                u.q[1] = *(const u32x4*)(p + 16);
                bfx[nt] = u.v;
            }
            #pragma unroll
            for (int mt = 0; mt < 4; ++mt)
                #pragma unroll
                for (int nt = 0; nt < 2; ++nt)
                    acc[mt][nt] = __builtin_amdgcn_wmma_f32_16x16x32_bf16(
                        false, af[mt], false, bfx[nt], (short)0, acc[mt][nt], false, false);
        }
    }

    // Epilogue: C layout = row (r + 8*half), col l16 within each 16x16 tile
    #pragma unroll
    for (int mt = 0; mt < 4; ++mt) {
        #pragma unroll
        for (int nt = 0; nt < 2; ++nt) {
            int gcol = n0 + wn * 32 + nt * 16 + l16;
            #pragma unroll
            for (int r = 0; r < 8; ++r) {
                int grow = m0 + wm * 64 + mt * 16 + r + 8 * half;
                storeC(C + (size_t)grow * N + gcol, acc[mt][nt][r]);
            }
        }
    }
}

// ---------------------------------------------------------------------------
// Flash attention: one block per (b, h, 128-row tile). Bc = 32 key columns.
// qkv: [ROWS][3*DIM] bf16 (q: col h*64, k: 1024+h*64, v: 2048+h*64)
// out: [ROWS][DIM] bf16
// ---------------------------------------------------------------------------
__global__ void __launch_bounds__(256) attn_kernel(const bf16* __restrict__ qkv,
                                                   bf16* __restrict__ out) {
    __shared__ __align__(16) bf16 Kt[32 * 64];      // [key][d]
    __shared__ __align__(16) bf16 Vt[64 * 32];      // [d][key] (transposed)
    __shared__ __align__(16) bf16 Pl[8][16 * 32];   // per-wave P patch [m][k]

    const int tid  = threadIdx.x;
    const int lane = tid & 31;
    const int wid  = tid >> 5;
    const int half = lane >> 4;
    const int l16  = lane & 15;

    const int blk = blockIdx.x;       // b*256 + h*16 + rb
    const int rb  = blk & 15;
    const int h   = (blk >> 4) & 15;
    const int b   = blk >> 8;

    const size_t rowbase = (size_t)b * SEQ;
    const int    trow0   = rb * 128;
    const float  scale   = 0.03125f;   // 1024^-0.5 (full model dim, per reference)

    // Q fragments for this wave's 16 rows, held in registers (A layout).
    v16bf qf[2];
    {
        const bf16* qrow = qkv + (rowbase + trow0 + wid * 16 + l16) * (3 * DIM) + h * DHEAD;
        #pragma unroll
        for (int s = 0; s < 2; ++s) {
            union { v16bf v; u32x4 q[2]; } u;
            u.q[0] = *(const u32x4*)(qrow + s * 32 + 8 * half);
            u.q[1] = *(const u32x4*)(qrow + s * 32 + 16 + 8 * half);
            qf[s] = u.v;
        }
    }

    float mrow[8], lrow[8];
    v8f acc[4];
    #pragma unroll
    for (int r = 0; r < 8; ++r) { mrow[r] = -3.0e38f; lrow[r] = 0.f; }
    #pragma unroll
    for (int nt = 0; nt < 4; ++nt) acc[nt] = (v8f){0.f,0.f,0.f,0.f,0.f,0.f,0.f,0.f};

    for (int j = 0; j < SEQ; j += 32) {
        // Stage K (row-major, async DMA) and V (transposed) tiles: 32x64 each
        {
            int r = tid >> 3, cc = (tid & 7) << 3;
            const bf16* base = qkv + (rowbase + j + r) * (3 * DIM) + h * DHEAD + cc;
            ASYNC_CP16(base + DIM, Kt + r * 64 + cc);
            u32x4 vd = *(const u32x4*)(base + 2 * DIM);
            const bf16* ve = (const bf16*)&vd;
            #pragma unroll
            for (int e = 0; e < 8; ++e) Vt[(cc + e) * 32 + r] = ve[e];
        }
        ASYNC_WAIT();
        __syncthreads();

        // S = Q @ K^T  (two 16x16 tiles covering 32 keys; K-dim = dh = 64)
        v8f sfrag[2];
        #pragma unroll
        for (int nt = 0; nt < 2; ++nt) {
            v8f c = (v8f){0.f,0.f,0.f,0.f,0.f,0.f,0.f,0.f};
            #pragma unroll
            for (int s = 0; s < 2; ++s) {
                const bf16* p = Kt + (nt * 16 + l16) * 64 + s * 32 + 8 * half;
                union { v16bf v; u32x4 q[2]; } u;
                u.q[0] = *(const u32x4*)p;
                u.q[1] = *(const u32x4*)(p + 16);
                c = __builtin_amdgcn_wmma_f32_16x16x32_bf16(
                        false, qf[s], false, u.v, (short)0, c, false, false);
            }
            sfrag[nt] = c;
        }

        // Online softmax. Row (r + 8*half) lives across the 16 lanes of this
        // lane-half -> shfl-xor masks 1,2,4,8 reduce it.
        float pv0[8], pv1[8], alpha[8];
        #pragma unroll
        for (int r = 0; r < 8; ++r) {
            float s0 = sfrag[0][r] * scale;
            float s1 = sfrag[1][r] * scale;
            float mx = fmaxf(s0, s1);
            #pragma unroll
            for (int off = 8; off >= 1; off >>= 1)
                mx = fmaxf(mx, __shfl_xor(mx, off, 32));
            float mnew = fmaxf(mrow[r], mx);
            float a  = __expf(mrow[r] - mnew);
            float p0 = __expf(s0 - mnew);
            float p1 = __expf(s1 - mnew);
            float rs = p0 + p1;
            #pragma unroll
            for (int off = 8; off >= 1; off >>= 1)
                rs += __shfl_xor(rs, off, 32);
            lrow[r] = lrow[r] * a + rs;
            mrow[r] = mnew;
            alpha[r] = a;
            pv0[r] = p0; pv1[r] = p1;
        }
        #pragma unroll
        for (int nt = 0; nt < 4; ++nt)
            #pragma unroll
            for (int r = 0; r < 8; ++r)
                acc[nt][r] *= alpha[r];

        // C-layout P -> A-layout P via per-wave LDS patch
        #pragma unroll
        for (int r = 0; r < 8; ++r) {
            Pl[wid][(r + 8 * half) * 32 + l16]      = (bf16)pv0[r];
            Pl[wid][(r + 8 * half) * 32 + 16 + l16] = (bf16)pv1[r];
        }
        union { v16bf v; u32x4 q[2]; } pu;
        {
            const bf16* pp = &Pl[wid][l16 * 32 + 8 * half];
            pu.q[0] = *(const u32x4*)pp;
            pu.q[1] = *(const u32x4*)(pp + 16);
        }

        // acc += P @ V   (k-dim = 32 keys, 4 n-tiles over dh = 64)
        #pragma unroll
        for (int nt = 0; nt < 4; ++nt) {
            const bf16* vp = Vt + (nt * 16 + l16) * 32 + 8 * half;
            union { v16bf v; u32x4 q[2]; } vu;
            vu.q[0] = *(const u32x4*)vp;
            vu.q[1] = *(const u32x4*)(vp + 16);
            acc[nt] = __builtin_amdgcn_wmma_f32_16x16x32_bf16(
                          false, pu.v, false, vu.v, (short)0, acc[nt], false, false);
        }
        __syncthreads();   // before next iteration overwrites Kt/Vt
    }

    // Normalize and write merged-head output (bf16) for the final GEMM.
    #pragma unroll
    for (int nt = 0; nt < 4; ++nt) {
        int col = h * DHEAD + nt * 16 + l16;
        #pragma unroll
        for (int r = 0; r < 8; ++r) {
            int row = trow0 + wid * 16 + r + 8 * half;
            out[(rowbase + row) * DIM + col] = (bf16)(acc[nt][r] / lrow[r]);
        }
    }
}

// ---------------------------------------------------------------------------
// Launch
// ---------------------------------------------------------------------------
extern "C" void kernel_launch(void* const* d_in, const int* in_sizes, int n_in,
                              void* d_out, int out_size, void* d_ws, size_t ws_size,
                              hipStream_t stream) {
    const float* x    = (const float*)d_in[0];  // [4,2048,1024]
    const float* wqkv = (const float*)d_in[1];  // [1024,3072]
    const float* wout = (const float*)d_in[2];  // [1024,1024]
    float* out = (float*)d_out;                 // [4,2048,1024]

    char* ws = (char*)d_ws;
    const size_t szXB   = (size_t)ROWS * DIM * 2;        // 16 MB
    const size_t szWQKV = (size_t)DIM * 3 * DIM * 2;     // 6 MB
    const size_t szWOUT = (size_t)DIM * DIM * 2;         // 2 MB
    const size_t szQKV  = (size_t)ROWS * 3 * DIM * 2;    // 48 MB
    bf16* xb    = (bf16*)(ws);
    bf16* wqkvb = (bf16*)(ws + szXB);
    bf16* woutb = (bf16*)(ws + szXB + szWQKV);
    bf16* qkvb  = (bf16*)(ws + szXB + szWQKV + szWOUT);
    bf16* aob   = (bf16*)(ws + szXB + szWQKV + szWOUT + szQKV);

    // 1) fp32 -> bf16 conversions
    {
        int n4 = ROWS * DIM / 4;
        cvt_f32_bf16<<<(n4 + 255) / 256, 256, 0, stream>>>((const v4f*)x, (v4bf*)xb, n4);
        n4 = DIM * 3 * DIM / 4;
        cvt_f32_bf16<<<(n4 + 255) / 256, 256, 0, stream>>>((const v4f*)wqkv, (v4bf*)wqkvb, n4);
        n4 = DIM * DIM / 4;
        cvt_f32_bf16<<<(n4 + 255) / 256, 256, 0, stream>>>((const v4f*)wout, (v4bf*)woutb, n4);
    }

    // 2) qkv = x @ W_qkv   [8192,1024] x [1024,3072] -> bf16 [8192,3072]
    gemm_bf16<bf16><<<(ROWS / 128) * (3 * DIM / 128), 256, 0, stream>>>(
        xb, wqkvb, qkvb, ROWS, 3 * DIM, DIM);

    // 3) attention -> merged heads bf16 [8192,1024]
    attn_kernel<<<BATCH * HEADS * (SEQ / 128), 256, 0, stream>>>(qkvb, aob);

    // 4) out = attn @ W_out  [8192,1024] x [1024,1024] -> f32
    gemm_bf16<float><<<(ROWS / 128) * (DIM / 128), 256, 0, stream>>>(
        aob, woutb, out, ROWS, DIM, DIM);
}
